// MultiHeadSelfAttention_171798692230
// MI455X (gfx1250) — compile-verified, matrix operand order checked by reading, not run on hardware
//
#include <hip/hip_runtime.h>
#include <hip/hip_bf16.h>

// ---------------------------------------------------------------------------
// MHA forward for MI455X (gfx1250, wave32, WMMA 16x16x32 f16 -> f32 acc)
//   B=2, S=2048, D=1024, H=16, hd=64
// ---------------------------------------------------------------------------

#define B_ 2
#define S_ 2048
#define D_ 1024
#define H_ 16
#define HD_ 64

typedef __attribute__((ext_vector_type(16))) _Float16 v16h;
typedef __attribute__((ext_vector_type(8)))  _Float16 v8h;
typedef __attribute__((ext_vector_type(8)))  float    v8f;

__device__ __forceinline__ v8f vsplat8(float x) {
  v8f v;
#pragma unroll
  for (int i = 0; i < 8; ++i) v[i] = x;
  return v;
}

// Load a 16x32 f16 tile (row-major, leading dim `ld` halfs) into the
// A-operand VGPR layout (ISA 7.12.2, 16-bit A 16x32). The same layout also
// serves as a B operand for the "N x K row-major" matrix, i.e. the WMMA then
// computes D = A * B_loaded^T  (exactly x @ W^T / Q @ K^T semantics).
//   lanes 0-15 : row = lane,    K chunks at {0..7}  and {16..23}
//   lanes 16-31: row = lane-16, K chunks at {8..15} and {24..31}
__device__ __forceinline__ v16h load_ab(const _Float16* p0, int ld, int lane) {
  const _Float16* p = p0 + (size_t)(lane & 15) * ld + ((lane >> 4) << 3);
  v8h lo = *(const v8h*)(p);
  v8h hi = *(const v8h*)(p + 16);
  return __builtin_shufflevector(lo, hi, 0, 1, 2, 3, 4, 5, 6, 7,
                                         8, 9, 10, 11, 12, 13, 14, 15);
}

__device__ __forceinline__ v8f wmma_f16(v16h a, v16h b, v8f c) {
  return __builtin_amdgcn_wmma_f32_16x16x32_f16(
      /*neg_a=*/false, a, /*neg_b=*/false, b,
      /*c_mod=*/(short)0, c, /*reuse_a=*/false, /*reuse_b=*/false);
}

// 16-lane-group reductions (rows of a C/D tile live across a 16-lane half;
// xor masks 1..8 never cross the half boundary in wave32).
__device__ __forceinline__ float gmax16(float v) {
  v = fmaxf(v, __shfl_xor(v, 1, 32));
  v = fmaxf(v, __shfl_xor(v, 2, 32));
  v = fmaxf(v, __shfl_xor(v, 4, 32));
  v = fmaxf(v, __shfl_xor(v, 8, 32));
  return v;
}
__device__ __forceinline__ float gsum16(float v) {
  v += __shfl_xor(v, 1, 32);
  v += __shfl_xor(v, 2, 32);
  v += __shfl_xor(v, 4, 32);
  v += __shfl_xor(v, 8, 32);
  return v;
}

// ---------------------------------------------------------------------------
__global__ __launch_bounds__(256) void f32_to_f16_kernel(
    const float* __restrict__ in, _Float16* __restrict__ out, int n) {
  int i = blockIdx.x * blockDim.x + threadIdx.x;
  int stride = gridDim.x * blockDim.x;
  for (; i < n; i += stride) out[i] = (_Float16)in[i];
}

// ---------------------------------------------------------------------------
// GEMM: out[M,N] = A[M,K] @ W[N,K]^T + bias[N]
// block = 256 threads = 8 waves; each wave computes a 32(M) x 64(N) tile:
// per K-step: 4 A-loads + 8 B-loads (b128) feed 8 WMMAs.
// grid.x = M/32, grid.y covers N in 512-column groups.
// ---------------------------------------------------------------------------
template <bool OUT_F16>
__global__ __launch_bounds__(256) void gemm_wmma_kernel(
    const _Float16* __restrict__ A, const _Float16* __restrict__ W,
    const float* __restrict__ bias, float* __restrict__ outF,
    _Float16* __restrict__ outH, int M, int N, int K) {
  const int lane = threadIdx.x & 31;
  // wave-uniform by construction; readfirstlane makes it scalar for codegen
  const int n0 =
      __builtin_amdgcn_readfirstlane((blockIdx.y * 8 + (threadIdx.x >> 5)) * 64);
  const int m0 = blockIdx.x * 32;
  if (n0 >= N) return;  // scalar branch
  const int col = lane & 15;
  const int mb = (lane >> 4) * 8;

  v8f acc[2][4];
#pragma unroll
  for (int j = 0; j < 4; ++j) {
    float bv = bias[n0 + j * 16 + col];
    acc[0][j] = vsplat8(bv);
    acc[1][j] = vsplat8(bv);
  }

  const _Float16* Ab0 = A + (size_t)m0 * K;
  const _Float16* Ab1 = A + (size_t)(m0 + 16) * K;
  for (int k0 = 0; k0 < K; k0 += 32) {
    __builtin_prefetch(Ab0 + k0 + 256, 0, 0);  // global_prefetch_b8
    __builtin_prefetch(Ab1 + k0 + 256, 0, 0);
    v16h a0 = load_ab(Ab0 + k0, K, lane);
    v16h a1 = load_ab(Ab1 + k0, K, lane);
#pragma unroll
    for (int j = 0; j < 4; ++j) {
      v16h b = load_ab(W + (size_t)(n0 + j * 16) * K + k0, K, lane);
      acc[0][j] = wmma_f16(a0, b, acc[0][j]);
      acc[1][j] = wmma_f16(a1, b, acc[1][j]);
    }
  }

#pragma unroll
  for (int mi = 0; mi < 2; ++mi) {
#pragma unroll
    for (int j = 0; j < 4; ++j) {
#pragma unroll
      for (int r = 0; r < 8; ++r) {
        size_t idx = (size_t)(m0 + mi * 16 + mb + r) * N + (n0 + j * 16 + col);
        if (OUT_F16) outH[idx] = (_Float16)acc[mi][j][r];
        else         outF[idx] = acc[mi][j][r];
      }
    }
  }
}

// ---------------------------------------------------------------------------
// Flash attention, causal. qkv is f16 [B,S,3D] (cols 0:1024=Q,1024:2048=K,
// 2048:3072=V; within each: head h -> cols h*64..h*64+63).
// One wave per 32-query tile (two stacked 16-row Q tiles share all K loads,
// the V transpose, and the V B-operands). 4 waves per block with private LDS
// slices; all wave indices go through readfirstlane so the causal key loop
// is a pure SALU loop and EXEC stays all-ones around every WMMA.
// ctx output: f16 [B,S,D].
// ---------------------------------------------------------------------------
__global__ __launch_bounds__(128) void attn_flash_kernel(
    const _Float16* __restrict__ qkv, _Float16* __restrict__ ctx) {
  __shared__ _Float16 lds[4][64 * 32 + 32 * 32];  // per wave: Vt[64][32], P[32][32]
  const int lane = threadIdx.x & 31;
  const int wave = __builtin_amdgcn_readfirstlane(threadIdx.x >> 5);
  _Float16* Vt = lds[wave];
  _Float16* Pl = Vt + 64 * 32;

  const int qi = __builtin_amdgcn_readfirstlane(blockIdx.x * 4 + wave);
  const int b  = qi >> 10;                // / (H * S/32)
  const int h  = (qi >> 6) & 15;
  const int s0 = (qi & 63) * 32;
  const int col = lane & 15;
  const int mb  = (lane >> 4) * 8;
  const size_t ldq = 3 * D_;

  // Two stacked Q tiles as A operands (hd=64 -> k-steps d0..31, d32..63)
  const _Float16* qbase = qkv + (size_t)(b * S_ + s0) * ldq + h * HD_;
  v16h qa[2][2];
  qa[0][0] = load_ab(qbase, ldq, lane);
  qa[0][1] = load_ab(qbase + 32, ldq, lane);
  qa[1][0] = load_ab(qbase + 16 * ldq, ldq, lane);
  qa[1][1] = load_ab(qbase + 16 * ldq + 32, ldq, lane);

  float mrun[2][8], lrun[2][8];
  v8f acc[2][4];
#pragma unroll
  for (int q = 0; q < 2; ++q) {
#pragma unroll
    for (int r = 0; r < 8; ++r) { mrun[q][r] = -3.0e38f; lrun[q][r] = 0.0f; }
#pragma unroll
    for (int j = 0; j < 4; ++j) acc[q][j] = vsplat8(0.0f);
  }

  const int ktmax = (s0 + 31) >> 5;  // inclusive, 32-key blocks; scalar
  for (int kt = 0; kt <= ktmax; ++kt) {
    const int t0 = kt * 32;

    // ---- scores for 32 keys x 32 queries: four 16x16 C tiles ----
    const _Float16* kbase = qkv + (size_t)(b * S_ + t0) * ldq + D_ + h * HD_;
    const v16h kb00 = load_ab(kbase, ldq, lane);             // keys 0..15, d0..31
    const v16h kb01 = load_ab(kbase + 32, ldq, lane);        // keys 0..15, d32..63
    const v16h kb10 = load_ab(kbase + 16 * ldq, ldq, lane);  // keys 16..31
    const v16h kb11 = load_ab(kbase + 16 * ldq + 32, ldq, lane);

    v8f sc[2][2];
#pragma unroll
    for (int q = 0; q < 2; ++q) {
      sc[q][0] = vsplat8(0.0f);
      sc[q][0] = wmma_f16(qa[q][0], kb00, sc[q][0]);
      sc[q][0] = wmma_f16(qa[q][1], kb01, sc[q][0]);
      sc[q][1] = vsplat8(0.0f);
      sc[q][1] = wmma_f16(qa[q][0], kb10, sc[q][1]);
      sc[q][1] = wmma_f16(qa[q][1], kb11, sc[q][1]);
    }

    // ---- V block (32 keys x 64) transposed into LDS: Vt[d][t] ----
    const _Float16* vrow =
        qkv + (size_t)(b * S_ + t0 + lane) * ldq + 2 * D_ + h * HD_;
#pragma unroll
    for (int c = 0; c < 8; ++c) {
      v8h vv = *(const v8h*)(vrow + c * 8);
#pragma unroll
      for (int i = 0; i < 8; ++i) Vt[(c * 8 + i) * 32 + lane] = vv[i];
    }

    // ---- online softmax (per tile-row; a row spans one 16-lane half) ----
#pragma unroll
    for (int q = 0; q < 2; ++q) {
      const int qrow0 = s0 + q * 16;
      if (t0 + 31 <= qrow0) {
        // fast path: block entirely at-or-below the diagonal, no mask
#pragma unroll
        for (int r = 0; r < 8; ++r) {
          float e0 = sc[q][0][r] * 0.125f;
          float e1 = sc[q][1][r] * 0.125f;
          float mnew = fmaxf(mrun[q][r], gmax16(fmaxf(e0, e1)));
          float scale = __expf(mrun[q][r] - mnew);
          float p0 = __expf(e0 - mnew);
          float p1 = __expf(e1 - mnew);
          lrun[q][r] = lrun[q][r] * scale + gsum16(p0 + p1);
          mrun[q][r] = mnew;
#pragma unroll
          for (int j = 0; j < 4; ++j) acc[q][j][r] *= scale;
          Pl[(q * 16 + mb + r) * 32 + col]      = (_Float16)p0;
          Pl[(q * 16 + mb + r) * 32 + col + 16] = (_Float16)p1;
        }
      } else {
#pragma unroll
        for (int r = 0; r < 8; ++r) {
          const int srow = qrow0 + mb + r;
          float e0 = sc[q][0][r] * 0.125f + ((t0 + col)      > srow ? -1.0e9f : 0.0f);
          float e1 = sc[q][1][r] * 0.125f + ((t0 + 16 + col) > srow ? -1.0e9f : 0.0f);
          float mnew = fmaxf(mrun[q][r], gmax16(fmaxf(e0, e1)));
          float scale = __expf(mrun[q][r] - mnew);
          float p0 = __expf(e0 - mnew);
          float p1 = __expf(e1 - mnew);
          lrun[q][r] = lrun[q][r] * scale + gsum16(p0 + p1);
          mrun[q][r] = mnew;
#pragma unroll
          for (int j = 0; j < 4; ++j) acc[q][j][r] *= scale;
          Pl[(q * 16 + mb + r) * 32 + col]      = (_Float16)p0;
          Pl[(q * 16 + mb + r) * 32 + col + 16] = (_Float16)p1;
        }
      }
    }

    // ---- context += P @ V  (A = P 16x32 per q-tile, B = Vt[d][t]) ----
    v16h pa0 = load_ab(Pl, 32, lane);
    v16h pa1 = load_ab(Pl + 16 * 32, 32, lane);
#pragma unroll
    for (int j = 0; j < 4; ++j) {
      v16h vb = load_ab(Vt + (size_t)(j * 16) * 32, 32, lane);
      acc[0][j] = wmma_f16(pa0, vb, acc[0][j]);
      acc[1][j] = wmma_f16(pa1, vb, acc[1][j]);
    }
  }

  // ---- normalize + store ctx f16 [B,S,D] ----
#pragma unroll
  for (int q = 0; q < 2; ++q) {
#pragma unroll
    for (int j = 0; j < 4; ++j) {
#pragma unroll
      for (int r = 0; r < 8; ++r) {
        size_t idx = (size_t)(b * S_ + s0 + q * 16 + mb + r) * D_
                   + h * HD_ + j * 16 + col;
        ctx[idx] = (_Float16)(acc[q][j][r] / lrun[q][r]);
      }
    }
  }
}

// ---------------------------------------------------------------------------
extern "C" void kernel_launch(void* const* d_in, const int* in_sizes, int n_in,
                              void* d_out, int out_size, void* d_ws, size_t ws_size,
                              hipStream_t stream) {
  (void)in_sizes; (void)n_in; (void)out_size; (void)ws_size;
  const float* x     = (const float*)d_in[0];
  // d_in[1] = attn_mask (causal, implemented analytically)
  const float* Wqkv  = (const float*)d_in[2];
  const float* bqkv  = (const float*)d_in[3];
  const float* Wout  = (const float*)d_in[4];
  const float* bout  = (const float*)d_in[5];
  float* out = (float*)d_out;

  const int nx    = B_ * S_ * D_;          // 4,194,304
  const int nqkv  = B_ * S_ * 3 * D_;      // 12,582,912
  const int nWqkv = 3 * D_ * D_;           // 3,145,728
  const int nWout = D_ * D_;               // 1,048,576

  _Float16* x16    = (_Float16*)d_ws;
  _Float16* qkv16  = x16 + nx;
  _Float16* ctx16  = qkv16 + nqkv;
  _Float16* Wqkv16 = ctx16 + nx;
  _Float16* Wout16 = Wqkv16 + nWqkv;       // total 48 MB of workspace

  // 1) fp32 -> f16 conversions
  f32_to_f16_kernel<<<2048, 256, 0, stream>>>(x, x16, nx);
  f32_to_f16_kernel<<<2048, 256, 0, stream>>>(Wqkv, Wqkv16, nWqkv);
  f32_to_f16_kernel<<<1024, 256, 0, stream>>>(Wout, Wout16, nWout);

  // 2) qkv = x @ Wqkv^T + bqkv  -> f16 [B*S, 3072]
  {
    dim3 grid(B_ * S_ / 32, 3 * D_ / 512);  // (128, 6)
    gemm_wmma_kernel<true><<<grid, 256, 0, stream>>>(
        x16, Wqkv16, bqkv, nullptr, qkv16, B_ * S_, 3 * D_, D_);
  }

  // 3) flash attention -> ctx f16 [B*S, 1024]
  {
    int qtiles = B_ * H_ * (S_ / 32);       // 2048 waves
    attn_flash_kernel<<<qtiles / 4, 128, 0, stream>>>(qkv16, ctx16);
  }

  // 4) out = ctx @ Wout^T + bout -> f32 [B*S, 1024]
  {
    dim3 grid(B_ * S_ / 32, D_ / 512);      // (128, 2)
    gemm_wmma_kernel<false><<<grid, 256, 0, stream>>>(
        ctx16, Wout16, bout, out, nullptr, B_ * S_, D_, D_);
  }
}
